// RED_DCM_Unet_16844861735430
// MI455X (gfx1250) — matrix operand order repaired and checked by reading, not compiled, not run on hardware
//
#include <hip/hip_runtime.h>
#include <cstdint>

// ---------------------------------------------------------------------------
// DualCrossMamba (RED_DCM_Unet) for gfx1250 / MI455X.
//  - all 4 projections via v_wmma_f32_16x16x32_bf16, K fully unrolled
//  - bf16 conversion done ONCE at producers (LN / conv / scan / weight-cvt)
//  - selective scan: 3-phase chunked scan; x_dbl tile staged to LDS via the
//    Tensor Data Mover (tensor_load_to_lds + s_wait_tensorcnt)
// ---------------------------------------------------------------------------

typedef __attribute__((ext_vector_type(16))) __bf16 v16bf;
typedef __attribute__((ext_vector_type(8)))  __bf16 bf8;
typedef __attribute__((ext_vector_type(8)))  float  v8f;
typedef unsigned int u32;
typedef __attribute__((ext_vector_type(4))) u32 v4u;
typedef __attribute__((ext_vector_type(8))) int v8i;
typedef __attribute__((ext_vector_type(4))) int v4i;

#define LSEQ   9216
#define DMODEL 96
#define DINNER 192
#define DSTATE 16
#define DTRANK 6
#define NCHUNK 96
#define CHLEN  96
#define XROWS  38          // dt_rank + 2*d_state rows of x_dbl

#ifndef __has_builtin
#define __has_builtin(x) 0
#endif
#if defined(__AMDGCN__) && __has_builtin(__builtin_amdgcn_tensor_load_to_lds) && \
    __has_builtin(__builtin_amdgcn_s_wait_tensorcnt)
#define HAVE_TDM 1
#else
#define HAVE_TDM 0
#endif

static __device__ __forceinline__ __bf16 f2bf(float f) {
    unsigned u = __builtin_bit_cast(unsigned, f);
    u += 0x7FFFu + ((u >> 16) & 1u);
    unsigned short h = (unsigned short)(u >> 16);
    return __builtin_bit_cast(__bf16, h);
}
static __device__ __forceinline__ float silu(float x) { return x / (1.f + __expf(-x)); }
static __device__ __forceinline__ float softplus_f(float x) {
    return x > 20.f ? x : log1pf(__expf(x));
}

// ---------------------------------------------------------------------------
// GEMM: C[MxN] = A[MxK](bf16,row-major) * W[NxK](bf16,row-major)^T, f32 acc.
// Fragment layouts per CDNA5 ISA 7.12.2 (wave32):
//   A: lane row m=lane&15, K = {kb..kb+7, kb+16..kb+23}, kb=8*(lane>>4)
//   B: lane col n=lane&15, K = {kb..kb+7, kb+16..kb+23}, kb=8*(lane>>4), same as A
// EPI==0: C[m*N+n];  EPI==1: C[n*M+m] = acc + resid[m*N+n]
// ---------------------------------------------------------------------------
template <int K, int EPI>
__global__ void k_gemm(const __bf16* __restrict__ A, int lda,
                       const __bf16* __restrict__ W, int ldw,
                       float* __restrict__ C, int M, int N,
                       const float* __restrict__ resid) {
    int wave = blockIdx.x * (blockDim.x >> 5) + (threadIdx.x >> 5);
    int lane = threadIdx.x & 31;
    int tilesM = M >> 4, tilesN = N >> 4;
    if (wave >= tilesM * tilesN) return;     // wave-uniform; EXEC stays all-ones
    int m0 = (wave % tilesM) << 4;
    int n0 = (wave / tilesM) << 4;
    int arow = m0 + (lane & 15);
    int akof = (lane >> 4) << 3;
    int brow = n0 + (lane & 15);
    int bkof = (lane >> 4) << 3;

    v8f acc = {};
#pragma unroll
    for (int k0 = 0; k0 < K; k0 += 32) {
        const bf8* pa = (const bf8*)(A + (long)arow * lda + k0 + akof);
        const bf8* pb = (const bf8*)(W + (long)brow * ldw + k0 + bkof);
        v16bf a = __builtin_shufflevector(pa[0], pa[2],
            0,1,2,3,4,5,6,7,8,9,10,11,12,13,14,15);
        v16bf b = __builtin_shufflevector(pb[0], pb[2],
            0,1,2,3,4,5,6,7,8,9,10,11,12,13,14,15);
        acc = __builtin_amdgcn_wmma_f32_16x16x32_bf16(
            false, a, false, b, (short)0, acc, false, false);
    }

    int n     = n0 + (lane & 15);
    int mbase = m0 + ((lane >> 4) << 3);
    if (EPI == 0) {
#pragma unroll
        for (int v = 0; v < 8; ++v)
            C[(long)(mbase + v) * N + n] = acc[v];
    } else {
#pragma unroll
        for (int v = 0; v < 8; ++v) {
            int m = mbase + v;
            C[(long)n * M + m] = acc[v] + resid[(long)m * N + n];
        }
    }
}

// ---------------------------------------------------------------------------
// LayerNorm: x (DMODEL, L) channel-major -> out (L, DMODEL); f32 and/or bf16.
// ---------------------------------------------------------------------------
__global__ void k_layernorm(const float* __restrict__ x,
                            const float* __restrict__ w,
                            const float* __restrict__ b,
                            float* __restrict__ outf,
                            __bf16* __restrict__ outbf, int L) {
    int l = blockIdx.x;
    int c = threadIdx.x;
    float v = x[(long)c * L + l];
    float s = v, sq = v * v;
#pragma unroll
    for (int off = 16; off; off >>= 1) {
        s  += __shfl_xor(s, off, 32);
        sq += __shfl_xor(sq, off, 32);
    }
    __shared__ float ss[3], ssq[3];
    if ((threadIdx.x & 31) == 0) {
        ss[threadIdx.x >> 5]  = s;
        ssq[threadIdx.x >> 5] = sq;
    }
    __syncthreads();
    float tot  = ss[0] + ss[1] + ss[2];
    float totq = ssq[0] + ssq[1] + ssq[2];
    float mu   = tot * (1.f / DMODEL);
    float var  = totq * (1.f / DMODEL) - mu * mu;
    float rstd = rsqrtf(var + 1e-5f);
    float r = (v - mu) * rstd * w[c] + b[c];
    if (outf)  outf [(long)l * DMODEL + c] = r;
    if (outbf) outbf[(long)l * DMODEL + c] = f2bf(r);
}

// ---------------------------------------------------------------------------
// Depthwise causal conv1d (k=4) + SiLU; writes f32 and/or bf16 (L, 192).
// ---------------------------------------------------------------------------
__global__ void k_conv_silu(const float* __restrict__ x, int stride,
                            const float* __restrict__ w,
                            const float* __restrict__ bias,
                            float* __restrict__ outf,
                            __bf16* __restrict__ outbf, int L) {
    int idx = blockIdx.x * blockDim.x + threadIdx.x;
    if (idx >= L * DINNER) return;
    int d = idx % DINNER;
    int l = idx / DINNER;
    float acc = bias[d];
#pragma unroll
    for (int i = 0; i < 4; ++i) {
        int ls = l - 3 + i;
        if (ls >= 0) acc += w[d * 4 + i] * x[(long)ls * stride + d];
    }
    float r = silu(acc);
    if (outf)  outf [(long)l * DINNER + d] = r;
    if (outbf) outbf[(long)l * DINNER + d] = f2bf(r);
}

// Weight convert (with zero pad): dst[i] = i<nsrc ? bf16(src[i]) : 0
__global__ void k_f32_to_bf16(const float* __restrict__ src,
                              __bf16* __restrict__ dst, int nsrc, int ntot) {
    int i = blockIdx.x * blockDim.x + threadIdx.x;
    if (i >= ntot) return;
    dst[i] = (i < nsrc) ? f2bf(src[i]) : f2bf(0.f);
}

// ---------------------------------------------------------------------------
// Stage the (XROWS x CHLEN) x_dbl tile for chunk c into LDS.
// TDM path: 2D tile descriptor (D#), data_size=4B, tensor (48 x L) row-major,
// tile (CHLEN wide x XROWS rows) -> packed row-major in LDS.
// ---------------------------------------------------------------------------
static __device__ __forceinline__ void stage_xdbl_tile(
        const float* __restrict__ xdbl, int c, int L, float* tile) {
#if HAVE_TDM
    if (threadIdx.x < 32) {                      // wave 0 issues the DMA
        unsigned long long ga = (unsigned long long)(uintptr_t)(xdbl + (long)c * CHLEN);
        u32 ldsOff = (u32)(uintptr_t)tile;       // low 32 bits of flat = LDS byte addr
        v4u g0 = { 1u,                            // count=1, user descriptor
                   ldsOff,                        // lds_addr
                   (u32)ga,                       // global_addr[31:0]
                   ((u32)(ga >> 32) & 0x01FFFFFFu) | (2u << 30) }; // addr[56:32] | type=2
        v8i g1 = { (int)(2u << 16),                               // data_size=4B
                   (int)(((u32)L & 0xFFFFu) << 16),               // tensor_dim0 lo16
                   (int)(((u32)L >> 16) | (48u << 16)),           // dim0 hi | tensor_dim1=48
                   (int)((u32)CHLEN << 16),                       // tile_dim0=96
                   (int)(u32)XROWS,                               // tile_dim1=38, tile_dim2=0
                   (int)(u32)L,                                   // tensor_dim0_stride lo32
                   0, 0 };
        v4i g2 = { 1, 1, 0, 0 };                 // tensor_dim2=1, tensor_dim3=1
        v4i g3 = { 0, 0x00010000, 0, 0 };        // tensor_dim4=1
#if defined(__clang_major__) && (__clang_major__ >= 23)
        v8i g4 = { 0, 0, 0, 0, 0, 0, 0, 0 };
        __builtin_amdgcn_tensor_load_to_lds(g0, g1, g2, g3, g4, 0);
#else
        __builtin_amdgcn_tensor_load_to_lds(g0, g1, g2, g3, 0);
#endif
        __builtin_amdgcn_s_wait_tensorcnt(0);
    }
    __syncthreads();
#else
    for (int i = threadIdx.x; i < XROWS * CHLEN; i += blockDim.x)
        tile[i] = xdbl[(long)(i / CHLEN) * L + c * CHLEN + (i % CHLEN)];
    __syncthreads();
#endif
}

// ---------------------------------------------------------------------------
// Selective scan, 3-phase chunked. tile rows: 0..5 dtr, 6..21 B, 22..37 C.
// ---------------------------------------------------------------------------
__global__ void k_scan_phase1(const float* __restrict__ xdbl,
                              const float* __restrict__ u,
                              const float* __restrict__ Alog,
                              const float* __restrict__ dtw,
                              const float* __restrict__ dtb,
                              float* __restrict__ hend,
                              float* __restrict__ dtsum, int L) {
    __shared__ float tile[XROWS * CHLEN];
    int c = blockIdx.x;
    int d = threadIdx.x;
    stage_xdbl_tile(xdbl, c, L, tile);

    float Av[DSTATE];
#pragma unroll
    for (int n = 0; n < DSTATE; ++n) Av[n] = -__expf(Alog[d * DSTATE + n]);
    float wv[DTRANK];
#pragma unroll
    for (int r = 0; r < DTRANK; ++r) wv[r] = dtw[d * DTRANK + r];
    float bias = dtb[d];
    float h[DSTATE];
#pragma unroll
    for (int n = 0; n < DSTATE; ++n) h[n] = 0.f;
    float dts = 0.f;
    int l0 = c * CHLEN;
    for (int t = 0; t < CHLEN; ++t) {
        float raw = bias;
#pragma unroll
        for (int r = 0; r < DTRANK; ++r) raw += wv[r] * tile[r * CHLEN + t];
        float dt = softplus_f(raw);
        float du = dt * u[(long)(l0 + t) * DINNER + d];
#pragma unroll
        for (int n = 0; n < DSTATE; ++n)
            h[n] = __expf(dt * Av[n]) * h[n] + du * tile[(DTRANK + n) * CHLEN + t];
        dts += dt;
    }
#pragma unroll
    for (int n = 0; n < DSTATE; ++n)
        hend[((long)c * DINNER + d) * DSTATE + n] = h[n];
    dtsum[c * DINNER + d] = dts;
}

__global__ void k_scan_phase2(const float* __restrict__ hend,
                              const float* __restrict__ dtsum,
                              const float* __restrict__ Alog,
                              float* __restrict__ prefix, int nchunk) {
    int d = threadIdx.x;
    float Av[DSTATE];
#pragma unroll
    for (int n = 0; n < DSTATE; ++n) Av[n] = -__expf(Alog[d * DSTATE + n]);
    float p[DSTATE];
#pragma unroll
    for (int n = 0; n < DSTATE; ++n) p[n] = 0.f;
    for (int c = 0; c < nchunk; ++c) {
        long base = ((long)c * DINNER + d) * DSTATE;
#pragma unroll
        for (int n = 0; n < DSTATE; ++n) prefix[base + n] = p[n];
        float ds = dtsum[c * DINNER + d];
#pragma unroll
        for (int n = 0; n < DSTATE; ++n)
            p[n] = __expf(Av[n] * ds) * p[n] + hend[base + n];
    }
}

__global__ void k_scan_phase3(const float* __restrict__ xdbl,
                              const float* __restrict__ u,
                              const float* __restrict__ xz,   // z gate source
                              const float* __restrict__ Alog,
                              const float* __restrict__ dtw,
                              const float* __restrict__ dtb,
                              const float* __restrict__ Dp,
                              const float* __restrict__ prefix,
                              __bf16* __restrict__ y, int L) {
    __shared__ float tile[XROWS * CHLEN];
    int c = blockIdx.x;
    int d = threadIdx.x;
    stage_xdbl_tile(xdbl, c, L, tile);

    float Av[DSTATE];
#pragma unroll
    for (int n = 0; n < DSTATE; ++n) Av[n] = -__expf(Alog[d * DSTATE + n]);
    float wv[DTRANK];
#pragma unroll
    for (int r = 0; r < DTRANK; ++r) wv[r] = dtw[d * DTRANK + r];
    float bias = dtb[d];
    float Dv = Dp[d];
    float h[DSTATE];
    long pbase = ((long)c * DINNER + d) * DSTATE;
#pragma unroll
    for (int n = 0; n < DSTATE; ++n) h[n] = prefix[pbase + n];
    int l0 = c * CHLEN;
    for (int t = 0; t < CHLEN; ++t) {
        int l = l0 + t;
        float raw = bias;
#pragma unroll
        for (int r = 0; r < DTRANK; ++r) raw += wv[r] * tile[r * CHLEN + t];
        float dt = softplus_f(raw);
        float uv = u[(long)l * DINNER + d];
        float du = dt * uv;
        float acc = Dv * uv;
#pragma unroll
        for (int n = 0; n < DSTATE; ++n) {
            h[n] = __expf(dt * Av[n]) * h[n] + du * tile[(DTRANK + n) * CHLEN + t];
            acc += h[n] * tile[(DTRANK + DSTATE + n) * CHLEN + t];
        }
        float z = xz[(long)l * (2 * DINNER) + DINNER + d];
        y[(long)l * DINNER + d] = f2bf(acc * silu(z));
    }
}

// ---------------------------------------------------------------------------
// Host side
// ---------------------------------------------------------------------------
extern "C" void kernel_launch(void* const* d_in, const int* in_sizes, int n_in,
                              void* d_out, int out_size, void* d_ws, size_t ws_size,
                              hipStream_t stream) {
    (void)in_sizes; (void)n_in; (void)out_size; (void)ws_size;
    const int L = LSEQ;
    const float* xin[2] = { (const float*)d_in[0], (const float*)d_in[1] };

    char* cur = (char*)d_ws;
    auto alloc = [&](size_t bytes) {
        char* p = cur; cur += (bytes + 255) & ~(size_t)255; return p;
    };
    float*  msln    = (float*) alloc((size_t)L * DMODEL * 4);
    __bf16* mslnbf  = (__bf16*)alloc((size_t)L * DMODEL * 2);
    __bf16* panlnbf = (__bf16*)alloc((size_t)L * DMODEL * 2);
    float*  xz      = (float*) alloc((size_t)L * 2 * DINNER * 4);
    float*  xb      = (float*) alloc((size_t)L * DINNER * 4);
    float*  uconv   = (float*) alloc((size_t)L * DINNER * 4);
    __bf16* xbconv  = (__bf16*)alloc((size_t)L * DINNER * 2);
    float*  xdbl    = (float*) alloc((size_t)48 * L * 4);
    __bf16* ybf     = (__bf16*)alloc((size_t)L * DINNER * 2);
    float*  hend    = (float*) alloc((size_t)NCHUNK * DINNER * DSTATE * 4);
    float*  dtsum   = (float*) alloc((size_t)NCHUNK * DINNER * 4);
    float*  prefix  = (float*) alloc((size_t)NCHUNK * DINNER * DSTATE * 4);
    __bf16* ipw_bf  = (__bf16*)alloc((size_t)384 * 96 * 2);
    __bf16* ipbw_bf = (__bf16*)alloc((size_t)192 * 96 * 2);
    __bf16* xpw_bf  = (__bf16*)alloc((size_t)48 * 192 * 2);
    __bf16* opw_bf  = (__bf16*)alloc((size_t)96 * 192 * 2);

    for (int br = 0; br < 2; ++br) {
        const float* ms  = xin[br];
        const float* pan = xin[1 - br];
        // per-branch params, jax dict (alphabetical) order
        void* const* P = d_in + 2 + br * 16;
        const float* Alog = (const float*)P[0];
        const float* Dp   = (const float*)P[1];
        const float* cbb  = (const float*)P[2];
        const float* cbw  = (const float*)P[3];
        const float* cb   = (const float*)P[4];
        const float* cw   = (const float*)P[5];
        const float* dtb  = (const float*)P[6];
        const float* dtw  = (const float*)P[7];
        const float* ipbw = (const float*)P[8];
        const float* ipw  = (const float*)P[9];
        const float* ln1b = (const float*)P[10];
        const float* ln1w = (const float*)P[11];
        const float* ln2b = (const float*)P[12];
        const float* ln2w = (const float*)P[13];
        const float* opw  = (const float*)P[14];
        const float* xpw  = (const float*)P[15];
        float* outp = (float*)d_out + (long)br * DMODEL * L;

        // weight f32 -> bf16 (x_proj zero-padded 38 -> 48 rows)
        k_f32_to_bf16<<<(384 * 96 + 255) / 256, 256, 0, stream>>>(ipw,  ipw_bf,  384 * 96, 384 * 96);
        k_f32_to_bf16<<<(192 * 96 + 255) / 256, 256, 0, stream>>>(ipbw, ipbw_bf, 192 * 96, 192 * 96);
        k_f32_to_bf16<<<(48 * 192 + 255) / 256, 256, 0, stream>>>(xpw,  xpw_bf,  XROWS * 192, 48 * 192);
        k_f32_to_bf16<<<(96 * 192 + 255) / 256, 256, 0, stream>>>(opw,  opw_bf,  96 * 192, 96 * 192);

        k_layernorm<<<L, DMODEL, 0, stream>>>(ms,  ln1w, ln1b, msln,    mslnbf,  L);
        k_layernorm<<<L, DMODEL, 0, stream>>>(pan, ln2w, ln2b, nullptr, panlnbf, L);

        // xz = msln @ in_proj_w^T : (L,384)
        { int w = (L >> 4) * (2 * DINNER >> 4);
          k_gemm<96, 0><<<(w + 3) / 4, 128, 0, stream>>>(mslnbf, DMODEL, ipw_bf, DMODEL, xz, L, 2 * DINNER, nullptr); }
        // x_b = panln @ in_proj_b_w[:192]^T : (L,192)
        { int w = (L >> 4) * (DINNER >> 4);
          k_gemm<96, 0><<<(w + 3) / 4, 128, 0, stream>>>(panlnbf, DMODEL, ipbw_bf, DMODEL, xb, L, DINNER, nullptr); }

        int cthreads = 256, cblocks = (L * DINNER + cthreads - 1) / cthreads;
        k_conv_silu<<<cblocks, cthreads, 0, stream>>>(xz, 2 * DINNER, cw,  cb,  uconv,  nullptr, L);
        k_conv_silu<<<cblocks, cthreads, 0, stream>>>(xb, DINNER,     cbw, cbb, nullptr, xbconv, L);

        // x_dbl = x_proj_w(38->48 x 192) @ xbconv^T : (48, L)
        { int w = (48 >> 4) * (L >> 4);
          k_gemm<192, 0><<<(w + 3) / 4, 128, 0, stream>>>(xpw_bf, DINNER, xbconv, DINNER, xdbl, 48, L, nullptr); }

        k_scan_phase1<<<NCHUNK, DINNER, 0, stream>>>(xdbl, uconv, Alog, dtw, dtb, hend, dtsum, L);
        k_scan_phase2<<<1, DINNER, 0, stream>>>(hend, dtsum, Alog, prefix, NCHUNK);
        k_scan_phase3<<<NCHUNK, DINNER, 0, stream>>>(xdbl, uconv, xz, Alog, dtw, dtb, Dp, prefix, ybf, L);

        // out = y @ out_proj_w^T + msln, stored transposed to (c, h, w)
        { int w = (L >> 4) * (DMODEL >> 4);
          k_gemm<192, 1><<<(w + 3) / 4, 128, 0, stream>>>(ybf, DINNER, opw_bf, DINNER, outp, L, DMODEL, msln); }
    }
}
